// Attention_63007170232645
// MI455X (gfx1250) — compile-verified, matrix operand order checked by reading, not run on hardware
//
#include <hip/hip_runtime.h>
#include <hip/hip_bf16.h>
#include <math.h>

// ---------------------------------------------------------------------------
// Problem constants (from reference)
// ---------------------------------------------------------------------------
#define BB   32     // batch
#define TQ   256    // decoder steps
#define TK   1024   // encoder steps
#define DD   512    // model dim
#define NH   4      // heads
#define PJ   512    // proj dim per head

typedef __bf16 bf16_t;
typedef __attribute__((ext_vector_type(16))) __bf16 v16bf;
typedef __attribute__((ext_vector_type(8)))  __bf16 bf8_t;
typedef __attribute__((ext_vector_type(8)))  float  v8f;
typedef __attribute__((ext_vector_type(4)))  int    v4i;

union FragU { v16bf f; bf8_t h[2]; };

#define LDS_AS __attribute__((address_space(3)))
#define GLB_AS __attribute__((address_space(1)))

// ---------------------------------------------------------------------------
// CDNA5 async global->LDS copy (ASYNCcnt-tracked), builtin preferred,
// inline-asm fallback. 16 bytes per lane.
// Builtin signature (probed from diagnostics): (v4i AS1* src, v4i AS3* dst,
// imm offset, imm cpol).
// ---------------------------------------------------------------------------
__device__ __forceinline__ void async_ld_b128(void* lds_dst, const void* gsrc) {
#if __has_builtin(__builtin_amdgcn_global_load_async_to_lds_b128)
    __builtin_amdgcn_global_load_async_to_lds_b128(
        (GLB_AS v4i*)gsrc, (LDS_AS v4i*)lds_dst, 0, 0);
#else
    asm volatile("global_load_async_to_lds_b128 %0, %1, off"
                 :
                 : "v"((unsigned)(unsigned long long)lds_dst),
                   "v"((unsigned long long)gsrc)
                 : "memory");
#endif
}

__device__ __forceinline__ void wait_async0() {
#if __has_builtin(__builtin_amdgcn_s_wait_asynccnt)
    __builtin_amdgcn_s_wait_asynccnt(0);
#else
    asm volatile("s_wait_asynccnt 0" ::: "memory");
#endif
}

// ---------------------------------------------------------------------------
// Small helper kernels
// ---------------------------------------------------------------------------
__global__ void bn_prep_kernel(const float* __restrict__ gamma,
                               const float* __restrict__ beta,
                               const float* __restrict__ mean,
                               const float* __restrict__ var,
                               float* __restrict__ scale,
                               float* __restrict__ bias, int n) {
    int i = blockIdx.x * blockDim.x + threadIdx.x;
    if (i < n) {
        float s = gamma[i] * rsqrtf(var[i] + 1e-5f);
        scale[i] = s;
        bias[i]  = beta[i] - mean[i] * s;
    }
}

__global__ void fill_unit_bn_kernel(float* __restrict__ ones,
                                    float* __restrict__ zeros, int n) {
    int i = blockIdx.x * blockDim.x + threadIdx.x;
    if (i < n) { ones[i] = 1.0f; zeros[i] = 0.0f; }
}

__global__ void f32_to_bf16_kernel(const float* __restrict__ x,
                                   bf16_t* __restrict__ y, long n) {
    long i = (long)blockIdx.x * blockDim.x + threadIdx.x;
    if (i < n) y[i] = (bf16_t)x[i];
}

// h: (B, TK, DD) f32  ->  ht: (B, DD, TK) bf16   (tiled LDS transpose)
__global__ __launch_bounds__(256)
void transpose_h_kernel(const float* __restrict__ h, bf16_t* __restrict__ ht) {
    __shared__ float tile[32][33];
    const int b  = blockIdx.z;
    const int k0 = blockIdx.x * 32;
    const int d0 = blockIdx.y * 32;
    const int tx = threadIdx.x;       // 0..31
    const int ty = threadIdx.y;       // 0..7
    const float* src = h + ((size_t)b * TK + k0) * DD + d0;
    #pragma unroll
    for (int i = 0; i < 32; i += 8)
        tile[ty + i][tx] = src[(size_t)(ty + i) * DD + tx];
    __syncthreads();
    bf16_t* dst = ht + ((size_t)b * DD + d0) * TK + k0;
    #pragma unroll
    for (int i = 0; i < 32; i += 8)
        dst[(size_t)(ty + i) * TK + tx] = (bf16_t)tile[tx][ty + i];
}

// Row-wise softmax over `cols` contiguous f32, one block per row.
__global__ __launch_bounds__(256)
void softmax_rows_kernel(float* __restrict__ a, int cols) {
    __shared__ float red[256];
    float* row = a + (size_t)blockIdx.x * cols;
    const int t = threadIdx.x;

    float m = -INFINITY;
    for (int c = t; c < cols; c += 256) m = fmaxf(m, row[c]);
    red[t] = m; __syncthreads();
    for (int s = 128; s > 0; s >>= 1) {
        if (t < s) red[t] = fmaxf(red[t], red[t + s]);
        __syncthreads();
    }
    m = red[0]; __syncthreads();

    float sum = 0.0f;
    for (int c = t; c < cols; c += 256) {
        float e = __expf(row[c] - m);
        row[c] = e;
        sum += e;
    }
    red[t] = sum; __syncthreads();
    for (int s = 128; s > 0; s >>= 1) {
        if (t < s) red[t] += red[t + s];
        __syncthreads();
    }
    const float inv = 1.0f / red[0];
    for (int c = t; c < cols; c += 256) row[c] *= inv;
}

// ---------------------------------------------------------------------------
// Tiled WMMA GEMM (NT form): C[z] = A[z] (MxK, row-major) * Bt[z]^T (Bt is NxK)
//   BN_A    : A is f32, apply per-column scale/bias while staging (else A bf16,
//             staged with async global->LDS copies)
//   OUT_BF16: write bf16 (else f32)
// Block: 256 threads = 8 waves; block tile 128x128; wave tile 32x64; K step 32.
// Double-buffered LDS: compute tile k while async-staging tile k+1.
// z decomposition: i1 = z / Z2, i2 = z % Z2; offsets = i1*s?1 + i2*s?2 (elems).
// ---------------------------------------------------------------------------
template <bool BN_A, bool OUT_BF16>
__global__ __launch_bounds__(256)
void gemm_nt_wmma(const void* __restrict__ Av,
                  const float* __restrict__ scale,
                  const float* __restrict__ bias,
                  const bf16_t* __restrict__ Bt,
                  void* __restrict__ Cv,
                  int M, int N, int K, int lda, int ldb, int ldc,
                  int Z2, long sA1, long sA2, long sB1, long sB2,
                  long sC1, long sC2) {
    constexpr int BM = 128, BNt = 128, BK = 32, PAD = 8, BKP = BK + PAD;
    __shared__ __align__(16) bf16_t As[2][BM][BKP];   // rows x K (K contiguous)
    __shared__ __align__(16) bf16_t Bs[2][BNt][BKP];  // cols x K (K contiguous)

    const int t     = threadIdx.x;
    const int lane  = t & 31;
    const int wid   = t >> 5;
    const int waveM = wid & 3;   // 0..3 -> 32 rows each
    const int waveN = wid >> 2;  // 0..1 -> 64 cols each

    const int  z  = blockIdx.z;
    const int  i1 = z / Z2;
    const int  i2 = z % Z2;
    const long offA = i1 * sA1 + i2 * sA2;
    const long offB = i1 * sB1 + i2 * sB2;
    const long offC = i1 * sC1 + i2 * sC2;

    const int rowBase = blockIdx.y * BM;
    const int colBase = blockIdx.x * BNt;

    const float*  Af = (const float*)Av  + offA;
    const bf16_t* Ab = (const bf16_t*)Av + offA;
    const bf16_t* Bp = Bt + offB;

    v8f acc[2][4];
    #pragma unroll
    for (int i = 0; i < 2; ++i)
        #pragma unroll
        for (int j = 0; j < 4; ++j)
            #pragma unroll
            for (int r = 0; r < 8; ++r) acc[i][j][r] = 0.0f;

    const int r0    = lane & 15;            // row/col within 16x16 tile
    const int khalf = (lane & 16) ? 8 : 0;  // ISA 16-bit A/B lane->K mapping

    // ---- staging: A tile (128 x 32) ----
    auto stageA = [&](int k0, int buf) {
        if (BN_A) {
            const int cg = t & 7;     // 8 groups of 4 floats = 32 cols
            const int rb = t >> 3;    // 0..31
            #pragma unroll
            for (int i = 0; i < 4; ++i) {
                const int r  = rb + 32 * i;
                const int kk = k0 + cg * 4;
                const float4 v =
                    *(const float4*)(Af + (long)(rowBase + r) * lda + kk);
                As[buf][r][cg * 4 + 0] = (bf16_t)(v.x * scale[kk + 0] + bias[kk + 0]);
                As[buf][r][cg * 4 + 1] = (bf16_t)(v.y * scale[kk + 1] + bias[kk + 1]);
                As[buf][r][cg * 4 + 2] = (bf16_t)(v.z * scale[kk + 2] + bias[kk + 2]);
                As[buf][r][cg * 4 + 3] = (bf16_t)(v.w * scale[kk + 3] + bias[kk + 3]);
            }
            if (k0 + BK < K)
                __builtin_prefetch(Af + (long)(rowBase + (t >> 3)) * lda + k0 + BK, 0, 1);
        } else {
            const int kg = t & 3;     // 4 groups of 8 bf16 = 32 cols
            const int rb = t >> 2;    // 0..63
            #pragma unroll
            for (int i = 0; i < 2; ++i) {
                const int r = rb + 64 * i;
                async_ld_b128(&As[buf][r][kg * 8],
                              Ab + (long)(rowBase + r) * lda + k0 + kg * 8);
            }
        }
    };
    // ---- staging: Bt tile (128 x 32), always bf16 -> async copy ----
    auto stageB = [&](int k0, int buf) {
        const int kg = t & 3;
        const int rb = t >> 2;        // 0..63
        #pragma unroll
        for (int i = 0; i < 2; ++i) {
            const int r = rb + 64 * i;
            async_ld_b128(&Bs[buf][r][kg * 8],
                          Bp + (long)(colBase + r) * ldb + k0 + kg * 8);
        }
    };
    // ---- wave MMA on one staged tile: 2x4 tiles of 16x16, K=32 ----
    auto compute = [&](int buf) {
        FragU a0, a1;
        const int ra = waveM * 32 + r0;
        a0.h[0] = *(const bf8_t*)&As[buf][ra][khalf];
        a0.h[1] = *(const bf8_t*)&As[buf][ra][16 + khalf];
        a1.h[0] = *(const bf8_t*)&As[buf][ra + 16][khalf];
        a1.h[1] = *(const bf8_t*)&As[buf][ra + 16][16 + khalf];
        #pragma unroll
        for (int j = 0; j < 4; ++j) {
            FragU bfr;
            const int c = waveN * 64 + j * 16 + r0;
            bfr.h[0] = *(const bf8_t*)&Bs[buf][c][khalf];
            bfr.h[1] = *(const bf8_t*)&Bs[buf][c][16 + khalf];
            acc[0][j] = __builtin_amdgcn_wmma_f32_16x16x32_bf16(
                false, a0.f, false, bfr.f, (short)0, acc[0][j], false, false);
            acc[1][j] = __builtin_amdgcn_wmma_f32_16x16x32_bf16(
                false, a1.f, false, bfr.f, (short)0, acc[1][j], false, false);
        }
    };

    // ---- software pipeline: double-buffered LDS ----
    const int nk = K / BK;
    stageA(0, 0);
    stageB(0, 0);
    wait_async0();
    __syncthreads();
    for (int kt = 0; kt < nk; ++kt) {
        const int cur = kt & 1;
        compute(cur);
        if (kt + 1 < nk) {
            stageA((kt + 1) * BK, cur ^ 1);
            stageB((kt + 1) * BK, cur ^ 1);
        }
        wait_async0();
        __syncthreads();
    }

    // ---- epilogue: C 16x16 f32 layout: elem r -> row r + (lane&16 ? 8 : 0),
    //      col = lane & 15 ----
    const int erow = (lane & 16) ? 8 : 0;
    #pragma unroll
    for (int i = 0; i < 2; ++i) {
        #pragma unroll
        for (int j = 0; j < 4; ++j) {
            const int rb = rowBase + waveM * 32 + i * 16 + erow;
            const int cb = colBase + waveN * 64 + j * 16 + (lane & 15);
            #pragma unroll
            for (int r = 0; r < 8; ++r) {
                const long idx = offC + (long)(rb + r) * ldc + cb;
                const float vv = acc[i][j][r];
                if (OUT_BF16) ((bf16_t*)Cv)[idx] = (bf16_t)vv;
                else          ((float*)Cv)[idx]  = vv;
            }
        }
    }
}

// ---------------------------------------------------------------------------
// Host-side orchestration
// ---------------------------------------------------------------------------
extern "C" void kernel_launch(void* const* d_in, const int* in_sizes, int n_in,
                              void* d_out, int out_size, void* d_ws, size_t ws_size,
                              hipStream_t stream) {
    (void)in_sizes; (void)n_in; (void)out_size; (void)ws_size;

    const float* s         = (const float*)d_in[0];   // (B, TQ, DD)
    const float* h         = (const float*)d_in[1];   // (B, TK, DD)
    const float* phi_gamma = (const float*)d_in[2];
    const float* phi_beta  = (const float*)d_in[3];
    const float* phi_mean  = (const float*)d_in[4];
    const float* phi_var   = (const float*)d_in[5];
    const float* W_phi     = (const float*)d_in[6];   // (NH*PJ, DD)
    const float* psi_gamma = (const float*)d_in[7];
    const float* psi_beta  = (const float*)d_in[8];
    const float* psi_mean  = (const float*)d_in[9];
    const float* psi_var   = (const float*)d_in[10];
    const float* W_psi     = (const float*)d_in[11];  // (PJ, DD)
    const float* red_gamma = (const float*)d_in[12];
    const float* red_beta  = (const float*)d_in[13];
    const float* red_mean  = (const float*)d_in[14];
    const float* red_var   = (const float*)d_in[15];
    const float* W_red     = (const float*)d_in[16];  // (DD, DD*NH)

    float* a_out = (float*)d_out;                            // (NH, B, TQ, TK)
    float* c_out = a_out + (size_t)NH * BB * TQ * TK;        // (B, TQ, DD)

    // ---- workspace carve-out ----
    char*  ws  = (char*)d_ws;
    size_t off = 0;
    auto take = [&](size_t bytes) -> char* {
        off = (off + 255) & ~(size_t)255;
        char* p = ws + off;
        off += bytes;
        return p;
    };
    float*  sc_phi = (float*)take(DD * 4);
    float*  bi_phi = (float*)take(DD * 4);
    float*  sc_psi = (float*)take(DD * 4);
    float*  bi_psi = (float*)take(DD * 4);
    float*  sc_red = (float*)take((size_t)DD * NH * 4);
    float*  bi_red = (float*)take((size_t)DD * NH * 4);
    float*  ones   = (float*)take((size_t)DD * NH * 4);
    float*  zeros  = (float*)take((size_t)DD * NH * 4);
    bf16_t* Wphi_b = (bf16_t*)take((size_t)NH * PJ * DD * 2);
    bf16_t* Wpsi_b = (bf16_t*)take((size_t)PJ * DD * 2);
    bf16_t* Wred_b = (bf16_t*)take((size_t)DD * DD * NH * 2);
    bf16_t* ms_b   = (bf16_t*)take((size_t)BB * TQ * NH * PJ * 2);   // (B*TQ, 2048)
    bf16_t* mh_b   = (bf16_t*)take((size_t)BB * TK * PJ * 2);        // (B*TK, 512)
    bf16_t* ht_b   = (bf16_t*)take((size_t)BB * DD * TK * 2);        // (B, DD, TK)
    float*  ctx_f  = (float*)take((size_t)BB * TQ * NH * DD * 4);    // (B*TQ, 2048)

    const dim3 blk256(256);

    // ---- 1. BN folds ----
    bn_prep_kernel<<<dim3((DD + 255) / 256), blk256, 0, stream>>>(
        phi_gamma, phi_beta, phi_mean, phi_var, sc_phi, bi_phi, DD);
    bn_prep_kernel<<<dim3((DD + 255) / 256), blk256, 0, stream>>>(
        psi_gamma, psi_beta, psi_mean, psi_var, sc_psi, bi_psi, DD);
    bn_prep_kernel<<<dim3((DD * NH + 255) / 256), blk256, 0, stream>>>(
        red_gamma, red_beta, red_mean, red_var, sc_red, bi_red, DD * NH);
    fill_unit_bn_kernel<<<dim3((DD * NH + 255) / 256), blk256, 0, stream>>>(
        ones, zeros, DD * NH);

    // ---- 2. weight conversion f32 -> bf16 ----
    {
        long n;
        n = (long)NH * PJ * DD;
        f32_to_bf16_kernel<<<dim3((unsigned)((n + 255) / 256)), blk256, 0, stream>>>(W_phi, Wphi_b, n);
        n = (long)PJ * DD;
        f32_to_bf16_kernel<<<dim3((unsigned)((n + 255) / 256)), blk256, 0, stream>>>(W_psi, Wpsi_b, n);
        n = (long)DD * DD * NH;
        f32_to_bf16_kernel<<<dim3((unsigned)((n + 255) / 256)), blk256, 0, stream>>>(W_red, Wred_b, n);
    }

    // ---- 3. transpose h -> (B, DD, TK) bf16 for the ctx GEMM ----
    transpose_h_kernel<<<dim3(TK / 32, DD / 32, BB), dim3(32, 8), 0, stream>>>(h, ht_b);

    // ---- 4. ms = BN(s) @ W_phi^T   (8192 x 2048, K=512) -> bf16 ----
    gemm_nt_wmma<true, true><<<dim3((NH * PJ) / 128, (BB * TQ) / 128, 1), blk256, 0, stream>>>(
        s, sc_phi, bi_phi, Wphi_b, ms_b,
        BB * TQ, NH * PJ, DD, DD, DD, NH * PJ,
        1, 0, 0, 0, 0, 0, 0);

    // ---- 5. mh = BN(h) @ W_psi^T   (32768 x 512, K=512) -> bf16 ----
    gemm_nt_wmma<true, true><<<dim3(PJ / 128, (BB * TK) / 128, 1), blk256, 0, stream>>>(
        h, sc_psi, bi_psi, Wpsi_b, mh_b,
        BB * TK, PJ, DD, DD, DD, PJ,
        1, 0, 0, 0, 0, 0, 0);

    // ---- 6. scores: e[h,b] = ms[b,:,h,:] @ mh[b]^T  (256 x 1024, K=512) -> a ----
    gemm_nt_wmma<false, false><<<dim3(TK / 128, TQ / 128, NH * BB), blk256, 0, stream>>>(
        ms_b, nullptr, nullptr, mh_b, a_out,
        TQ, TK, PJ, NH * PJ, PJ, TK,
        BB,
        /*sA1(head)*/ (long)PJ,            /*sA2(batch)*/ (long)TQ * NH * PJ,
        /*sB1*/ 0,                         /*sB2*/ (long)TK * PJ,
        /*sC1*/ (long)BB * TQ * TK,        /*sC2*/ (long)TQ * TK);

    // ---- 7. softmax over Tk, in place in a ----
    softmax_rows_kernel<<<dim3(NH * BB * TQ), blk256, 0, stream>>>(a_out, TK);

    // ---- 8. ctx[h,b] = a[h,b] @ h[b]  (256 x 512, K=1024) -> ctx_f (f32) ----
    gemm_nt_wmma<true, false><<<dim3(DD / 128, TQ / 128, NH * BB), blk256, 0, stream>>>(
        a_out, ones, zeros, ht_b, ctx_f,
        TQ, DD, TK, TK, TK, NH * DD,
        BB,
        /*sA1*/ (long)BB * TQ * TK,        /*sA2*/ (long)TQ * TK,
        /*sB1*/ 0,                         /*sB2*/ (long)DD * TK,
        /*sC1*/ (long)DD,                  /*sC2*/ (long)TQ * NH * DD);

    // ---- 9. c = BN(ctx) @ W_red^T  (8192 x 512, K=2048) -> c_out ----
    gemm_nt_wmma<true, false><<<dim3(DD / 128, (BB * TQ) / 128, 1), blk256, 0, stream>>>(
        ctx_f, sc_red, bi_red, Wred_b, c_out,
        BB * TQ, DD, NH * DD, NH * DD, NH * DD, DD,
        1, 0, 0, 0, 0, 0, 0);
}
